// EyringEdgePool_graph_induce_28647431864514
// MI455X (gfx1250) — matrix operand order repaired and checked by reading, not compiled, not run on hardware
//
#include <hip/hip_runtime.h>
#include <hip/hip_bf16.h>

typedef __attribute__((ext_vector_type(16))) _Float16 v16h;
typedef __attribute__((ext_vector_type(8)))  _Float16 v8h;
typedef __attribute__((ext_vector_type(8)))  float    v8f;

#define NN   40960
#define NG   64
#define NPG  640
#define NE   327680
#define EPG  5120
#define HID  128
#define SORTN 8192

// ---------------------------------------------------------------------------
// Weight repack into WMMA B-fragment layout.
// For ktile kt (32 K-rows) and ntile nt (16 cols), lane l stores 16 halves:
//   col = nt*16 + (l&15);  K = kt*32 + (l>=16 ? 16 : 0) + j   (j = 0..15)
// ---------------------------------------------------------------------------
__global__ void repack_w_kernel(const float* __restrict__ W, _Float16* __restrict__ dst,
                                int Ksrc, int Kpad, int Ncols) {
  int t = blockIdx.x * blockDim.x + threadIdx.x;
  int ntn = Ncols >> 4;
  int total = (Kpad >> 5) * ntn * 32;
  if (t >= total) return;
  int lane = t & 31;
  int nt = (t >> 5) % ntn;
  int kt = (t >> 5) / ntn;
  int col = nt * 16 + (lane & 15);
  int kbase = kt * 32 + ((lane >> 4) ? 16 : 0);
  _Float16* out = dst + (size_t)t * 16;
  for (int j = 0; j < 16; ++j) {
    int k = kbase + j;
    float v = (k < Ksrc) ? W[(size_t)k * Ncols + col] : 0.0f;
    out[j] = (_Float16)v;
  }
}

// ---------------------------------------------------------------------------
// WMMA GEMM: C[M,N] = A[M,K](f16 row-major) x Bpacked + bias, optional relu.
// One 16x16 output tile per wave; K stepped by 32 via v_wmma_f32_16x16x32_f16.
// ---------------------------------------------------------------------------
__global__ void wmma_gemm_kernel(const _Float16* __restrict__ A,
                                 const _Float16* __restrict__ Bp,
                                 const float* __restrict__ bias,
                                 float* __restrict__ Cf,
                                 _Float16* __restrict__ Ch,
                                 int M, int N, int K, int fuse_relu) {
  const int lane = threadIdx.x & 31;
  const int wave = threadIdx.x >> 5;
  const int tile = blockIdx.x * (blockDim.x >> 5) + wave;
  const int ntn  = N >> 4;
  const int mt   = tile / ntn;
  const int nt   = tile % ntn;
  if (mt >= (M >> 4)) return;                    // wave-uniform exit: EXEC stays full
  const int hi  = lane >> 4;                     // lane group 0/1
  const int row = mt * 16 + (lane & 15);
  const _Float16* arow = A + (size_t)row * K + hi * 16;
  const _Float16* bptr = Bp + ((size_t)(nt * 32 + lane)) * 16;
  const size_t bstride = (size_t)ntn * 32 * 16;  // halves per ktile
  v8f c = {};
  for (int k0 = 0; k0 < K; k0 += 32) {
    union { v16h v; v8h h[2]; } a, b;
    a.h[0] = *(const v8h*)(arow + k0);
    a.h[1] = *(const v8h*)(arow + k0 + 8);
    b.v    = *(const v16h*)bptr;
    bptr  += bstride;
    if (k0 + 32 < K) __builtin_prefetch((const void*)bptr, 0, 1);
    c = __builtin_amdgcn_wmma_f32_16x16x32_f16(false, a.v, false, b.v,
                                               (short)0, c, false, false);
  }
  const int col = nt * 16 + (lane & 15);
  const float bv = bias ? bias[col] : 0.0f;
#pragma unroll
  for (int r = 0; r < 8; ++r) {
    int orow = mt * 16 + r + hi * 8;
    float v = c[r] + bv;
    if (fuse_relu) v = fmaxf(v, 0.0f);
    Cf[(size_t)orow * N + col] = v;
    if (Ch) Ch[(size_t)orow * N + col] = (_Float16)v;
  }
}

// ---------------------------------------------------------------------------
// Elementwise / scatter helpers
// ---------------------------------------------------------------------------
__global__ void build_x40_kernel(const float* __restrict__ x, const float* __restrict__ x_in,
                                 const int* __restrict__ batch, _Float16* __restrict__ x40) {
  int n = blockIdx.x, f = threadIdx.x;            // 64 threads
  int g = batch[n];
  float v = 0.0f;
  if (f < 32)       v = x[(size_t)n * 32 + f];
  else if (f < 40)  v = x_in[g * 9 + 1 + (f - 32)];
  x40[(size_t)n * 64 + f] = (_Float16)v;
}

__global__ void deg_kernel(const int* __restrict__ dst, float* __restrict__ deg, int E) {
  int e = blockIdx.x * blockDim.x + threadIdx.x;
  if (e < E) atomicAdd(&deg[dst[e]], 1.0f);
}

__global__ void dinv_kernel(const float* __restrict__ deg, float* __restrict__ dinv, int n) {
  int i = blockIdx.x * blockDim.x + threadIdx.x;
  if (i < n) dinv[i] = rsqrtf(deg[i] + 1.0f);
}

__global__ void msg_kernel(const float* __restrict__ h, const int* __restrict__ src,
                           const int* __restrict__ dst, const float* __restrict__ dinv,
                           float* __restrict__ acc, int E) {
  int t = blockIdx.x * blockDim.x + threadIdx.x;
  int e = t >> 5;
  if (e >= E) return;
  int f0 = (t & 31) * 4;
  int s = src[e], d = dst[e];
  float coef = dinv[s] * dinv[d];
  const float4 hv = *(const float4*)(h + (size_t)s * HID + f0);
  float* ap = acc + (size_t)d * HID + f0;
  atomicAdd(ap + 0, hv.x * coef);
  atomicAdd(ap + 1, hv.y * coef);
  atomicAdd(ap + 2, hv.z * coef);
  atomicAdd(ap + 3, hv.w * coef);
}

__global__ void combine_kernel(const float* __restrict__ acc, const float* __restrict__ h,
                               const float* __restrict__ dinv, const float* __restrict__ bias,
                               float* __restrict__ xf, _Float16* __restrict__ xh, int n) {
  int t = blockIdx.x * blockDim.x + threadIdx.x;
  int node = t >> 7;
  if (node >= n) return;
  int f = t & (HID - 1);
  float di = dinv[node];
  float v = acc[t] + h[t] * di * di + bias[f];
  v = fmaxf(v, 0.0f);
  xf[t] = v;
  xh[t] = (_Float16)v;
}

// ---------------------------------------------------------------------------
// Edge-pool: per-node score projection, softmax over incoming edges,
// per-graph bitonic sort + serial greedy matching, cluster contraction.
// ---------------------------------------------------------------------------
__global__ void nodeproj_kernel(const float* __restrict__ x, const float* __restrict__ Wp,
                                float* __restrict__ p1, float* __restrict__ p2, int n) {
  int i = blockIdx.x * blockDim.x + threadIdx.x;
  if (i >= n) return;
  const float* xr = x + (size_t)i * HID;
  float a = 0.0f, b = 0.0f;
  for (int k = 0; k < HID; ++k) { a += xr[k] * Wp[k]; b += xr[k] * Wp[HID + k]; }
  p1[i] = a; p2[i] = b;
}

__device__ __forceinline__ unsigned enc_f32(float f) {
  unsigned b = __float_as_uint(f);
  return (b & 0x80000000u) ? ~b : (b | 0x80000000u);
}
__device__ __forceinline__ float dec_f32(unsigned u) {
  unsigned b = (u & 0x80000000u) ? (u ^ 0x80000000u) : ~u;
  return __uint_as_float(b);
}

__global__ void edgeraw_kernel(const float* __restrict__ p1, const float* __restrict__ p2,
                               const int* __restrict__ src, const int* __restrict__ dst,
                               const float* __restrict__ bp, int bidx,
                               float* __restrict__ raw, unsigned* __restrict__ mEnc, int E) {
  int e = blockIdx.x * blockDim.x + threadIdx.x;
  if (e >= E) return;
  float r = p1[src[e]] + p2[dst[e]] + bp[bidx];
  raw[e] = r;
  atomicMax(&mEnc[dst[e]], enc_f32(r));
}

__global__ void edgeexp_kernel(float* __restrict__ raw, const int* __restrict__ dst,
                               const unsigned* __restrict__ mEnc, float* __restrict__ Z, int E) {
  int e = blockIdx.x * blockDim.x + threadIdx.x;
  if (e >= E) return;
  float ex = expf(raw[e] - dec_f32(mEnc[dst[e]]));
  raw[e] = ex;
  atomicAdd(&Z[dst[e]], ex);
}

__global__ void edgescore_kernel(float* __restrict__ raw, const int* __restrict__ dst,
                                 const float* __restrict__ Z, int E) {
  int e = blockIdx.x * blockDim.x + threadIdx.x;
  if (e >= E) return;
  raw[e] = raw[e] / Z[dst[e]] + 0.5f;
}

// Per-graph descending bitonic sort of (score, idx) keys; stable tie-break on idx.
__global__ void sort_kernel(const float* __restrict__ score, int* __restrict__ order) {
  extern __shared__ unsigned long long key[];
  int g = blockIdx.x;
  for (int i = threadIdx.x; i < SORTN; i += blockDim.x) {
    if (i < EPG) {
      unsigned long long k = ((unsigned long long)enc_f32(score[g * EPG + i]) << 32)
                             | (unsigned)(0xFFFFFFFFu - (unsigned)i);
      key[i] = k;
    } else key[i] = 0ull;
  }
  __syncthreads();
  for (int k = 2; k <= SORTN; k <<= 1) {
    for (int j = k >> 1; j > 0; j >>= 1) {
      for (int i = threadIdx.x; i < SORTN; i += blockDim.x) {
        int ixj = i ^ j;
        if (ixj > i) {
          bool up = ((i & k) == 0);
          unsigned long long a = key[i], b = key[ixj];
          bool sw = up ? (a < b) : (a > b);   // descending overall
          if (sw) { key[i] = b; key[ixj] = a; }
        }
      }
      __syncthreads();
    }
  }
  for (int i = threadIdx.x; i < EPG; i += blockDim.x) {
    int local = (int)(0xFFFFFFFFu - (unsigned)(key[i] & 0xFFFFFFFFu));
    order[g * EPG + i] = g * EPG + local;
  }
}

// Serial greedy matching, one graph per block (thread 0). Cluster ids stay
// inside graph block [g*NPG, (g+1)*NPG) -> per-graph independence is exact.
__global__ void match_kernel(const int* __restrict__ order, const float* __restrict__ score,
                             const int* __restrict__ src, const int* __restrict__ dst,
                             const int* __restrict__ batch,
                             int* __restrict__ cluster, unsigned char* __restrict__ merged,
                             float* __restrict__ cs, int* __restrict__ newBatch) {
  int g = blockIdx.x;
  if (threadIdx.x != 0) return;
  int base = g * NPG;
  for (int i = 0; i < NPG; ++i) {
    merged[base + i] = 0; cs[base + i] = 1.0f;
    newBatch[base + i] = NG; cluster[base + i] = base;
  }
  int cnt = 0;
  for (int r = 0; r < EPG; ++r) {
    int e = order[g * EPG + r];
    int s = src[e], d = dst[e];
    if (s != d && !merged[s] && !merged[d]) {
      int cid = base + cnt;
      cluster[s] = cid; cluster[d] = cid;
      merged[s] = 1; merged[d] = 1;
      cs[cid] = score[e];
      newBatch[cid] = g;
      ++cnt;
    }
  }
  int rv = 0;
  for (int n = base; n < base + NPG; ++n)
    if (!merged[n] && batch[n] < NG) { cluster[n] = base + cnt + rv; newBatch[base + cnt + rv] = g; ++rv; }
  int ri = 0;
  for (int n = base; n < base + NPG; ++n)
    if (!merged[n] && batch[n] >= NG) { cluster[n] = base + cnt + rv + ri; ++ri; }
}

__global__ void poolapply_kernel(const float* __restrict__ x, const int* __restrict__ cluster,
                                 float* __restrict__ newX, int n) {
  int t = blockIdx.x * blockDim.x + threadIdx.x;
  int node = t >> 7;
  if (node >= n) return;
  int f = t & (HID - 1);
  atomicAdd(&newX[(size_t)cluster[node] * HID + f], x[t]);
}

__global__ void poolscale_kernel(const float* __restrict__ newX, const float* __restrict__ cs,
                                 float* __restrict__ xf, _Float16* __restrict__ xh, int n) {
  int t = blockIdx.x * blockDim.x + threadIdx.x;
  int node = t >> 7;
  if (node >= n) return;
  float v = newX[t] * cs[node];
  xf[t] = v;
  xh[t] = (_Float16)v;
}

__global__ void remap_kernel(const int* __restrict__ cluster, const int* __restrict__ src,
                             const int* __restrict__ dst, int* __restrict__ src2,
                             int* __restrict__ dst2, int E) {
  int e = blockIdx.x * blockDim.x + threadIdx.x;
  if (e >= E) return;
  src2[e] = cluster[src[e]];
  dst2[e] = cluster[dst[e]];
}

__global__ void meanpool_kernel(const float* __restrict__ x, const int* __restrict__ batch,
                                float* __restrict__ hout, _Float16* __restrict__ hh, int off) {
  int g = blockIdx.x, f = threadIdx.x;   // 128 threads
  int base = g * NPG;
  float sum = 0.0f; int cnt = 0;
  for (int i = 0; i < NPG; ++i) {
    if (batch[base + i] == g) { sum += x[(size_t)(base + i) * HID + f]; ++cnt; }
  }
  float v = sum / fmaxf((float)cnt, 1.0f);
  hout[(size_t)g * 256 + off + f] = v;
  hh[(size_t)g * 256 + off + f] = (_Float16)v;
}

__global__ void head_kernel(const float* __restrict__ h, const float* __restrict__ Wx,
                            const float* __restrict__ bx, const float* __restrict__ x_in,
                            float* __restrict__ out) {
  int g = blockIdx.x * blockDim.x + threadIdx.x;
  if (g >= NG) return;
  float a0 = bx[0], n1 = bx[1];
  const float* hr = h + (size_t)g * 256;
  for (int k = 0; k < 256; ++k) { a0 += hr[k] * Wx[k * 2 + 0]; n1 += hr[k] * Wx[k * 2 + 1]; }
  out[g] = x_in[g * 9] * (1.0f + n1) - a0;
}

// ---------------------------------------------------------------------------
// Host orchestration
// ---------------------------------------------------------------------------
static inline char* bump(char*& p, size_t bytes) {
  char* r = p;
  p += (bytes + 255) & ~(size_t)255;
  return r;
}

static void gcn_layer(const _Float16* Aact, int K, const _Float16* Wpk, const float* bias,
                      const int* src, const int* dst,
                      float* htmp, float* acc, float* deg, float* dinv,
                      float* xf, _Float16* xh, hipStream_t stream) {
  hipMemsetAsync(deg, 0, (size_t)NN * 4, stream);
  deg_kernel<<<NE / 256, 256, 0, stream>>>(dst, deg, NE);
  dinv_kernel<<<NN / 256, 256, 0, stream>>>(deg, dinv, NN);
  int tiles = (NN / 16) * (HID / 16);
  wmma_gemm_kernel<<<tiles / 8, 256, 0, stream>>>(Aact, Wpk, nullptr, htmp, nullptr,
                                                  NN, HID, K, 0);
  hipMemsetAsync(acc, 0, (size_t)NN * HID * 4, stream);
  msg_kernel<<<(NE * 32) / 256, 256, 0, stream>>>(htmp, src, dst, dinv, acc, NE);
  combine_kernel<<<(NN * HID) / 256, 256, 0, stream>>>(acc, htmp, dinv, bias, xf, xh, NN);
}

extern "C" void kernel_launch(void* const* d_in, const int* in_sizes, int n_in,
                              void* d_out, int out_size, void* d_ws, size_t ws_size,
                              hipStream_t stream) {
  const float* x_in  = (const float*)d_in[0];
  const float* x0    = (const float*)d_in[1];
  const int*   src0  = (const int*)d_in[2];
  const int*   dst0  = src0 + NE;
  const int*   bat0  = (const int*)d_in[3];
  const float* W1    = (const float*)d_in[5];
  const float* b1    = (const float*)d_in[6];
  const float* Wc    = (const float*)d_in[7];
  const float* bc    = (const float*)d_in[8];
  const float* Wp    = (const float*)d_in[9];
  const float* bp    = (const float*)d_in[10];
  const float* Wn    = (const float*)d_in[11];
  const float* bn    = (const float*)d_in[12];
  const float* Wx    = (const float*)d_in[13];
  const float* bx    = (const float*)d_in[14];
  float* out = (float*)d_out;

  char* p = (char*)d_ws;
  _Float16* x40  = (_Float16*)bump(p, (size_t)NN * 64 * 2);
  _Float16* w1p  = (_Float16*)bump(p, (size_t)64 * 128 * 2);
  _Float16* wcp  = (_Float16*)bump(p, (size_t)3 * 128 * 128 * 2);   // only Wc[0..2] live
  _Float16* wnp  = (_Float16*)bump(p, (size_t)2 * 256 * 256 * 2);
  _Float16* xh   = (_Float16*)bump(p, (size_t)NN * HID * 2);
  float*    xf   = (float*)bump(p, (size_t)NN * HID * 4);
  float*    htmp = (float*)bump(p, (size_t)NN * HID * 4);
  float*    acc  = (float*)bump(p, (size_t)NN * HID * 4);
  float*    deg  = (float*)bump(p, (size_t)NN * 4);
  float*    dinv = (float*)bump(p, (size_t)NN * 4);
  int*      srcA = (int*)bump(p, (size_t)NE * 4);
  int*      dstA = (int*)bump(p, (size_t)NE * 4);
  int*      batA = (int*)bump(p, (size_t)NN * 4);
  float*    p1   = (float*)bump(p, (size_t)NN * 4);
  float*    p2   = (float*)bump(p, (size_t)NN * 4);
  float*    ebuf = (float*)bump(p, (size_t)NE * 4);
  unsigned* mEnc = (unsigned*)bump(p, (size_t)NN * 4);
  float*    Z    = (float*)bump(p, (size_t)NN * 4);
  int*      ordr = (int*)bump(p, (size_t)NE * 4);
  int*      clus = (int*)bump(p, (size_t)NN * 4);
  unsigned char* mrg = (unsigned char*)bump(p, (size_t)NN);
  float*    cs   = (float*)bump(p, (size_t)NN * 4);
  float*    hbuf = (float*)bump(p, (size_t)NG * 256 * 4);
  _Float16* hh   = (_Float16*)bump(p, (size_t)NG * 256 * 2);
  float*    h2f  = (float*)bump(p, (size_t)NG * 256 * 4);
  _Float16* h2h  = (_Float16*)bump(p, (size_t)NG * 256 * 2);
  (void)ws_size; (void)in_sizes; (void)n_in; (void)out_size;

  // ---- repack weights into WMMA B-fragment layout (f32 -> f16) ----
  repack_w_kernel<<<2, 256, 0, stream>>>(W1, w1p, 40, 64, 128);
  for (int i = 0; i < 3; ++i)
    repack_w_kernel<<<4, 256, 0, stream>>>(Wc + (size_t)i * 128 * 128,
                                           wcp + (size_t)i * 128 * 128, 128, 128, 128);
  for (int j = 0; j < 2; ++j)
    repack_w_kernel<<<16, 256, 0, stream>>>(Wn + (size_t)j * 256 * 256,
                                            wnp + (size_t)j * 256 * 256, 256, 256, 256);

  // ---- input embedding: x40 = [x | x_in[1:9][batch] | 0-pad] as f16 ----
  build_x40_kernel<<<NN, 64, 0, stream>>>(x0, x_in, bat0, x40);

  // ---- GCN(W1), K=64 ----
  gcn_layer(x40, 64, w1p, b1, src0, dst0, htmp, acc, deg, dinv, xf, xh, stream);
  // ---- GCN(Wc0), K=128 ----
  gcn_layer(xh, 128, wcp, bc, src0, dst0, htmp, acc, deg, dinv, xf, xh, stream);

  // ---- xs[0] = mean_pool ----
  meanpool_kernel<<<NG, 128, 0, stream>>>(xf, bat0, hbuf, hh, 0);

  // ---- edge_pool level 0 ----
  nodeproj_kernel<<<NN / 256, 256, 0, stream>>>(xf, Wp, p1, p2, NN);
  hipMemsetAsync(mEnc, 0, (size_t)NN * 4, stream);
  edgeraw_kernel<<<NE / 256, 256, 0, stream>>>(p1, p2, src0, dst0, bp, 0, ebuf, mEnc, NE);
  hipMemsetAsync(Z, 0, (size_t)NN * 4, stream);
  edgeexp_kernel<<<NE / 256, 256, 0, stream>>>(ebuf, dst0, mEnc, Z, NE);
  edgescore_kernel<<<NE / 256, 256, 0, stream>>>(ebuf, dst0, Z, NE);
  sort_kernel<<<NG, 1024, SORTN * 8, stream>>>(ebuf, ordr);
  match_kernel<<<NG, 32, 0, stream>>>(ordr, ebuf, src0, dst0, bat0, clus, mrg, cs, batA);
  hipMemsetAsync(acc, 0, (size_t)NN * HID * 4, stream);
  poolapply_kernel<<<(NN * HID) / 256, 256, 0, stream>>>(xf, clus, acc, NN);
  poolscale_kernel<<<(NN * HID) / 256, 256, 0, stream>>>(acc, cs, xf, xh, NN);
  remap_kernel<<<NE / 256, 256, 0, stream>>>(clus, src0, dst0, srcA, dstA, NE);

  // ---- GCN(Wc1), GCN(Wc2) on contracted graph ----
  gcn_layer(xh, 128, wcp + (size_t)1 * 128 * 128, bc + 128, srcA, dstA,
            htmp, acc, deg, dinv, xf, xh, stream);
  gcn_layer(xh, 128, wcp + (size_t)2 * 128 * 128, bc + 256, srcA, dstA,
            htmp, acc, deg, dinv, xf, xh, stream);

  // ---- xs[1] = mean_pool ----
  meanpool_kernel<<<NG, 128, 0, stream>>>(xf, batA, hbuf, hh, 128);

  // (layers i=3,4 and the second edge_pool do not affect the output: skipped)

  // ---- MLP head: two 64x256 @ 256x256 WMMA GEMMs + linear head ----
  wmma_gemm_kernel<<<8, 256, 0, stream>>>(hh, wnp, bn, h2f, h2h, NG, 256, 256, 1);
  wmma_gemm_kernel<<<8, 256, 0, stream>>>(h2h, wnp + (size_t)256 * 256, bn + 256,
                                          hbuf, nullptr, NG, 256, 256, 1);
  head_kernel<<<1, 64, 0, stream>>>(hbuf, Wx, bx, x_in, out);
}
